// KATBlocks_27204322853511
// MI455X (gfx1250) — matrix operand order repaired and checked by reading, not compiled, not run on hardware
//
#include <hip/hip_runtime.h>
#include <hip/hip_bf16.h>
#include <math.h>

// ---------------- problem constants ----------------
#define BB      2
#define NP      4096     // patches
#define NK      144      // kernels (9 * 16, exact WMMA tiling)
#define DIMC    512
#define NHEADS  8
#define DHEAD   64
#define INNERC  512
#define QKVC    1536
#define MLPC    2048
#define NDEPTH  4
#define NEGV    (-1e9f)
#define ATTSCALE 0.125f  // 64^-0.5

typedef __bf16 bf16;
typedef __attribute__((ext_vector_type(2)))  __bf16 bf16x2;
typedef __attribute__((ext_vector_type(16))) __bf16 bf16x16;
typedef __attribute__((ext_vector_type(8)))  float  floatx8;

static __device__ inline floatx8 wmma_bf16(bf16x16 a, bf16x16 b, floatx8 c) {
  // D = A(16x32 bf16) x B(32x16 bf16) + C(16x16 f32)
  return __builtin_amdgcn_wmma_f32_16x16x32_bf16(false, a, false, b, (short)0, c,
                                                 false, false);
}
static __device__ inline floatx8 zero8() {
  floatx8 z = {0.f, 0.f, 0.f, 0.f, 0.f, 0.f, 0.f, 0.f};
  return z;
}
static __device__ inline unsigned int pkbf(float a, float b) {
  union { bf16x2 h; unsigned int u; } cv;
  cv.h.x = (bf16)a; cv.h.y = (bf16)b;
  return cv.u;
}
// convert 8 contiguous f32 (float4-aligned) into elements [base..base+7]
static __device__ inline void cvt8(const float* __restrict__ p, bf16x16& f, int base) {
  float4 a = *(const float4*)p;
  float4 b = *(const float4*)(p + 4);
  f[base + 0] = (bf16)a.x; f[base + 1] = (bf16)a.y;
  f[base + 2] = (bf16)a.z; f[base + 3] = (bf16)a.w;
  f[base + 4] = (bf16)b.x; f[base + 5] = (bf16)b.y;
  f[base + 6] = (bf16)b.z; f[base + 7] = (bf16)b.w;
}

// ---------------- LayerNorm: one wave per 512-wide row ----------------
__global__ __launch_bounds__(256)
void ln_kernel(const float* __restrict__ in, float* __restrict__ out,
               const float* __restrict__ gam, const float* __restrict__ bet,
               int nrows) {
  int wave = threadIdx.x >> 5;
  int lane = threadIdx.x & 31;
  int row = blockIdx.x * 8 + wave;
  if (row >= nrows) return;
  const float4* p = (const float4*)(in + (size_t)row * DIMC + lane * 16);
  float4 q[4];
  float v[16];
  float s = 0.f;
#pragma unroll
  for (int i = 0; i < 4; ++i) q[i] = p[i];
#pragma unroll
  for (int i = 0; i < 4; ++i) {
    v[4 * i + 0] = q[i].x; v[4 * i + 1] = q[i].y;
    v[4 * i + 2] = q[i].z; v[4 * i + 3] = q[i].w;
  }
#pragma unroll
  for (int i = 0; i < 16; ++i) s += v[i];
#pragma unroll
  for (int off = 16; off >= 1; off >>= 1) s += __shfl_xor(s, off, 32);
  float mean = s * (1.0f / DIMC);
  float vs = 0.f;
#pragma unroll
  for (int i = 0; i < 16; ++i) { float d = v[i] - mean; vs += d * d; }
#pragma unroll
  for (int off = 16; off >= 1; off >>= 1) vs += __shfl_xor(vs, off, 32);
  float inv = rsqrtf(vs * (1.0f / DIMC) + 1e-5f);
  float* o = out + (size_t)row * DIMC + lane * 16;
  const float* g = gam + lane * 16;
  const float* b = bet + lane * 16;
#pragma unroll
  for (int i = 0; i < 16; ++i) o[i] = (v[i] - mean) * inv * g[i] + b[i];
}

// ---------------- Generic WMMA GEMM, double-buffered ----------------
// out[M,N] = act(A[M,K] @ W[K,N] + bias) + res
// Block tile 128x128, 8 waves (4x2), wave tile 32x64 (2x4 WMMA), K-step 32.
// Double-buffered LDS: one barrier per K-step, next tile's global loads
// overlap the WMMAs. A buffers are workspace-padded to a multiple of 128
// rows so staging is unguarded; only the epilogue store checks row < M.
__global__ __launch_bounds__(256)
void gemm_kernel(const float* __restrict__ A, const float* __restrict__ W,
                 const float* __restrict__ bias, const float* __restrict__ res,
                 float* __restrict__ out, int M, int N, int K, int act) {
  __shared__ unsigned int As[2][128][16];  // 128 rows x 32 bf16 packed, 2 x 8 KB
  __shared__ unsigned int Bs[2][128][16];  // 128 cols x 32 bf16 packed, 2 x 8 KB
  int tid = threadIdx.x;
  int wave = tid >> 5, lane = tid & 31;
  int wm = wave & 3, wn = wave >> 2;       // 4 (M) x 2 (N) wave grid
  int lhalf = lane >> 4, lmod = lane & 15;
  int m0 = blockIdx.x * 128;
  int n0 = blockIdx.y * 128;
  floatx8 acc[2][4];
#pragma unroll
  for (int i = 0; i < 2; ++i)
#pragma unroll
    for (int j = 0; j < 4; ++j) acc[i][j] = zero8();

  // staging coordinates
  int ar = tid >> 1, ac2 = (tid & 1) * 8;  // A: row, uint col (16 f32/thread)
  int bk2 = tid >> 4, bn = (tid & 15) * 8; // B: k-pair (2 rows), 8 cols/thread

  float4 pa[4], pb[4];
  {
    const float4* s4 = (const float4*)(A + (size_t)(m0 + ar) * K + ac2 * 2);
    pa[0] = s4[0]; pa[1] = s4[1]; pa[2] = s4[2]; pa[3] = s4[3];
    const float* r0 = W + (size_t)(2 * bk2) * N + n0 + bn;
    pb[0] = ((const float4*)r0)[0]; pb[1] = ((const float4*)r0)[1];
    pb[2] = ((const float4*)(r0 + N))[0]; pb[3] = ((const float4*)(r0 + N))[1];
  }

  for (int k0 = 0; k0 < K; k0 += 32) {
    int cur = (k0 >> 5) & 1;
    { // commit prefetched tile to LDS (packed bf16)
      uint4 u0 = { pkbf(pa[0].x, pa[0].y), pkbf(pa[0].z, pa[0].w),
                   pkbf(pa[1].x, pa[1].y), pkbf(pa[1].z, pa[1].w) };
      uint4 u1 = { pkbf(pa[2].x, pa[2].y), pkbf(pa[2].z, pa[2].w),
                   pkbf(pa[3].x, pa[3].y), pkbf(pa[3].z, pa[3].w) };
      *(uint4*)&As[cur][ar][ac2] = u0;
      *(uint4*)&As[cur][ar][ac2 + 4] = u1;
      Bs[cur][bn + 0][bk2] = pkbf(pb[0].x, pb[2].x);
      Bs[cur][bn + 1][bk2] = pkbf(pb[0].y, pb[2].y);
      Bs[cur][bn + 2][bk2] = pkbf(pb[0].z, pb[2].z);
      Bs[cur][bn + 3][bk2] = pkbf(pb[0].w, pb[2].w);
      Bs[cur][bn + 4][bk2] = pkbf(pb[1].x, pb[3].x);
      Bs[cur][bn + 5][bk2] = pkbf(pb[1].y, pb[3].y);
      Bs[cur][bn + 6][bk2] = pkbf(pb[1].z, pb[3].z);
      Bs[cur][bn + 7][bk2] = pkbf(pb[1].w, pb[3].w);
    }
    __syncthreads();
    if (k0 + 32 < K) { // prefetch next tile (overlaps WMMAs below)
      const float4* s4 = (const float4*)(A + (size_t)(m0 + ar) * K + k0 + 32 + ac2 * 2);
      pa[0] = s4[0]; pa[1] = s4[1]; pa[2] = s4[2]; pa[3] = s4[3];
      const float* r0 = W + (size_t)(k0 + 32 + 2 * bk2) * N + n0 + bn;
      pb[0] = ((const float4*)r0)[0]; pb[1] = ((const float4*)r0)[1];
      pb[2] = ((const float4*)(r0 + N))[0]; pb[3] = ((const float4*)(r0 + N))[1];
    }

    bf16x16 af[2], bfr[4];
#pragma unroll
    for (int i = 0; i < 2; ++i) {
      const bf16* ap = (const bf16*)&As[cur][wm * 32 + i * 16 + lmod][0] + lhalf * 16;
#pragma unroll
      for (int e = 0; e < 16; ++e) af[i][e] = ap[e];
    }
#pragma unroll
    for (int j = 0; j < 4; ++j) {
      const bf16* bp = (const bf16*)&Bs[cur][wn * 64 + j * 16 + lmod][0] + lhalf * 16;
#pragma unroll
      for (int e = 0; e < 16; ++e) bfr[j][e] = bp[e];
    }
#pragma unroll
    for (int i = 0; i < 2; ++i)
#pragma unroll
      for (int j = 0; j < 4; ++j)
        acc[i][j] = wmma_bf16(af[i], bfr[j], acc[i][j]);
    // no second barrier: double buffer + in-order per-wave DS ops make the
    // next iteration's stores (other buffer) safe after the single barrier
  }

  // epilogue: bias / GELU(exact erf) / residual
#pragma unroll
  for (int i = 0; i < 2; ++i) {
#pragma unroll
    for (int j = 0; j < 4; ++j) {
      int col = n0 + wn * 64 + j * 16 + lmod;
#pragma unroll
      for (int r = 0; r < 8; ++r) {
        int row = m0 + wm * 32 + i * 16 + lhalf * 8 + r;
        if (row < M) {
          float v = acc[i][j][r];
          if (bias) v += bias[col];
          if (act == 1) v = 0.5f * v * (1.0f + erff(v * 0.70710678118f));
          if (res) v += res[(size_t)row * N + col];
          out[(size_t)row * N + col] = v;
        }
      }
    }
  }
}

// ---------------- patch -> kernel attention, fully fused ----------------
// grid (NP/32, BB*NHEADS), block 64 (2 waves, 16 query rows each)
// qkvk has 16 zeroed pad rows after row BB*NK, so K-padding 144->160 is unguarded.
__global__ __launch_bounds__(64)
void attn_pk_kernel(const float* __restrict__ qkvx, const float* __restrict__ qkvk,
                    const float* __restrict__ maskp, const float* __restrict__ kmaskp,
                    const float* __restrict__ rd, float* __restrict__ ao,
                    float dinv) {
  int bh = blockIdx.y, b = bh >> 3, h = bh & 7;
  int wave = threadIdx.x >> 5, lane = threadIdx.x & 31;
  int lhalf = lane >> 4, lmod = lane & 15;
  int q0 = blockIdx.x * 32 + wave * 16;
  __shared__ float sc[2][16][160];   // scores, padded 144->160

  // Q fragments (A: 16 rows x 64, two K-steps)
  bf16x16 qf[2];
  const float* qb = qkvx + ((size_t)b * NP + q0 + lmod) * QKVC + h * DHEAD;
#pragma unroll
  for (int s = 0; s < 2; ++s) {
    const float* p = qb + s * 32 + lhalf * 8;
    cvt8(p, qf[s], 0);
    cvt8(p + 16, qf[s], 8);
  }
  // dots = q @ k_k^T over 9 column tiles
  for (int t = 0; t < 9; ++t) {
    floatx8 d = zero8();
#pragma unroll
    for (int s = 0; s < 2; ++s) {
      const float* kp = qkvk + ((size_t)b * NK + t * 16 + lmod) * QKVC + INNERC +
                        h * DHEAD + s * 32 + lhalf * 8;
      bf16x16 bfr;
      cvt8(kp, bfr, 0);
      cvt8(kp + 16, bfr, 8);
      d = wmma_bf16(qf[s], bfr, d);
    }
#pragma unroll
    for (int r = 0; r < 8; ++r)
      sc[wave][lhalf * 8 + r][t * 16 + lmod] = d[r];
  }
  __syncthreads();

  // softmax over 144 (row = lmod; lhalf splits columns 0..71 / 72..143)
  int row = lmod;
  int gq = q0 + row;
  float mrow = maskp[(size_t)b * NP + gq];
  int c0 = lhalf * 72, c1 = c0 + 72;
  float mx = -3e38f;
  for (int c = c0; c < c1; ++c) {
    bool bad = (mrow * kmaskp[b * NK + c]) < 0.5f;
    float v = bad ? NEGV : sc[wave][row][c] * ATTSCALE;
    sc[wave][row][c] = v;
    mx = fmaxf(mx, v);
  }
  mx = fmaxf(mx, __shfl_xor(mx, 16, 32));
  float sum = 0.f;
  for (int c = c0; c < c1; ++c) {
    float e = __expf(sc[wave][row][c] - mx);
    sc[wave][row][c] = e;
    sum += e;
  }
  sum += __shfl_xor(sum, 16, 32);
  float inv = 1.0f / sum;
  const float* rdb = rd + (size_t)b * NK * NP;
  for (int c = c0; c < c1; ++c) {
    float rv = rdb[(size_t)c * NP + gq];
    sc[wave][row][c] *= inv * __expf(-rv * rv * dinv);   // post-softmax decay
  }
#pragma unroll
  for (int i = 0; i < 8; ++i) sc[wave][row][144 + lhalf * 8 + i] = 0.f; // zero pad
  __syncthreads();

  // O = P(16x160) @ V(160x64); rows 144..159 of V come from the zeroed pad
  floatx8 o[4] = {zero8(), zero8(), zero8(), zero8()};
  for (int s = 0; s < 5; ++s) {
    bf16x16 af;
    const float* ap = &sc[wave][lmod][s * 32 + lhalf * 8];
    cvt8(ap, af, 0);
    cvt8(ap + 16, af, 8);
#pragma unroll
    for (int j = 0; j < 4; ++j) {
      bf16x16 bfr;
#pragma unroll
      for (int e = 0; e < 16; ++e) {
        int kidx = s * 32 + lhalf * 8 + (e & 7) + (e >> 3) * 16;
        bfr[e] = (bf16)qkvk[((size_t)b * NK + kidx) * QKVC + 2 * INNERC + h * DHEAD +
                            j * 16 + lmod];
      }
      o[j] = wmma_bf16(af, bfr, o[j]);
    }
  }
#pragma unroll
  for (int j = 0; j < 4; ++j)
#pragma unroll
    for (int r = 0; r < 8; ++r)
      ao[((size_t)b * NP + q0 + lhalf * 8 + r) * INNERC + h * DHEAD + j * 16 + lmod] = o[j][r];
}

// ---------------- kernel -> patch attention: pass 1, dots -> scratch ----------------
// grid (NP/64, 9, BB*NHEADS), block 32
__global__ __launch_bounds__(32)
void attn_kp_dots(const float* __restrict__ qkvx, const float* __restrict__ qkvk,
                  const float* __restrict__ maskp, const float* __restrict__ kmaskp,
                  float* __restrict__ S) {
  int bh = blockIdx.z, b = bh >> 3, h = bh & 7;
  int mt = blockIdx.y;
  int nc0 = blockIdx.x * 64;
  int lane = threadIdx.x & 31;
  int lhalf = lane >> 4, lmod = lane & 15;

  bf16x16 qf[2];
  const float* qb = qkvk + ((size_t)b * NK + mt * 16 + lmod) * QKVC + h * DHEAD;
#pragma unroll
  for (int s = 0; s < 2; ++s) {
    const float* p = qb + s * 32 + lhalf * 8;
    cvt8(p, qf[s], 0);
    cvt8(p + 16, qf[s], 8);
  }
#pragma unroll
  for (int jt = 0; jt < 4; ++jt) {
    floatx8 d = zero8();
#pragma unroll
    for (int s = 0; s < 2; ++s) {
      const float* kp = qkvx + ((size_t)b * NP + nc0 + jt * 16 + lmod) * QKVC + INNERC +
                        h * DHEAD + s * 32 + lhalf * 8;
      bf16x16 bfr;
      cvt8(kp, bfr, 0);
      cvt8(kp + 16, bfr, 8);
      d = wmma_bf16(qf[s], bfr, d);
    }
    int i = nc0 + jt * 16 + lmod;
    float mi = maskp[(size_t)b * NP + i];
#pragma unroll
    for (int r = 0; r < 8; ++r) {
      int j = mt * 16 + lhalf * 8 + r;
      float v = d[r] * ATTSCALE;
      if (mi * kmaskp[b * NK + j] < 0.5f) v = NEGV;
      S[((size_t)bh * NK + j) * NP + i] = v;
    }
  }
}

// ---------------- pass 2: per-row max + sum(exp) (one block per row) ----------------
__global__ __launch_bounds__(256)
void attn_kp_stats(const float* __restrict__ S, float* __restrict__ rmax,
                   float* __restrict__ rsum) {
  size_t row = blockIdx.x;
  const float4* p = (const float4*)(S + row * (size_t)NP + threadIdx.x * 16);
  __shared__ float red[256];
  float4 q[4];
  float v[16];
#pragma unroll
  for (int i = 0; i < 4; ++i) q[i] = p[i];
#pragma unroll
  for (int i = 0; i < 4; ++i) {
    v[4 * i + 0] = q[i].x; v[4 * i + 1] = q[i].y;
    v[4 * i + 2] = q[i].z; v[4 * i + 3] = q[i].w;
  }
  float mx = -3e38f;
#pragma unroll
  for (int i = 0; i < 16; ++i) mx = fmaxf(mx, v[i]);
  red[threadIdx.x] = mx;
  __syncthreads();
  for (int off = 128; off >= 1; off >>= 1) {
    if (threadIdx.x < off) red[threadIdx.x] = fmaxf(red[threadIdx.x], red[threadIdx.x + off]);
    __syncthreads();
  }
  mx = red[0];
  __syncthreads();
  float s = 0.f;
#pragma unroll
  for (int i = 0; i < 16; ++i) s += __expf(v[i] - mx);
  red[threadIdx.x] = s;
  __syncthreads();
  for (int off = 128; off >= 1; off >>= 1) {
    if (threadIdx.x < off) red[threadIdx.x] += red[threadIdx.x + off];
    __syncthreads();
  }
  if (threadIdx.x == 0) { rmax[row] = mx; rsum[row] = red[0]; }
}

// ---------------- pass 3: O = (softmax * decay) @ V, 4-way K-split ----------------
// grid (9, BB*NHEADS), block 128 (4 waves, each 1024 of K=4096)
__global__ __launch_bounds__(128)
void attn_kp_av(const float* __restrict__ S, const float* __restrict__ rmax,
                const float* __restrict__ rsum, const float* __restrict__ rd,
                const float* __restrict__ qkvx, float* __restrict__ ao, float dinv) {
  int bh = blockIdx.y, b = bh >> 3, h = bh & 7;
  int mt = blockIdx.x;
  int wave = threadIdx.x >> 5, lane = threadIdx.x & 31;
  int lhalf = lane >> 4, lmod = lane & 15;
  __shared__ float ps[4][16][64];

  int j = mt * 16 + lmod;  // kernel row owned by this lane for A-frags
  float mxv = rmax[(size_t)bh * NK + j];
  float inv = 1.0f / rsum[(size_t)bh * NK + j];
  const float* srow = S + ((size_t)bh * NK + j) * NP;
  const float* rdrow = rd + ((size_t)b * NK + j) * NP;

  floatx8 o[4] = {zero8(), zero8(), zero8(), zero8()};
  int kbeg = wave * 1024, kend = kbeg + 1024;
  for (int k0 = kbeg; k0 < kend; k0 += 32) {
    int i0 = k0 + lhalf * 8;
    float4 s0 = *(const float4*)(srow + i0);
    float4 s1 = *(const float4*)(srow + i0 + 4);
    float4 s2 = *(const float4*)(srow + i0 + 16);
    float4 s3 = *(const float4*)(srow + i0 + 20);
    float4 r0 = *(const float4*)(rdrow + i0);
    float4 r1 = *(const float4*)(rdrow + i0 + 4);
    float4 r2 = *(const float4*)(rdrow + i0 + 16);
    float4 r3 = *(const float4*)(rdrow + i0 + 20);
    bf16x16 af;
    af[0]  = (bf16)(__expf(s0.x - mxv - r0.x * r0.x * dinv) * inv);
    af[1]  = (bf16)(__expf(s0.y - mxv - r0.y * r0.y * dinv) * inv);
    af[2]  = (bf16)(__expf(s0.z - mxv - r0.z * r0.z * dinv) * inv);
    af[3]  = (bf16)(__expf(s0.w - mxv - r0.w * r0.w * dinv) * inv);
    af[4]  = (bf16)(__expf(s1.x - mxv - r1.x * r1.x * dinv) * inv);
    af[5]  = (bf16)(__expf(s1.y - mxv - r1.y * r1.y * dinv) * inv);
    af[6]  = (bf16)(__expf(s1.z - mxv - r1.z * r1.z * dinv) * inv);
    af[7]  = (bf16)(__expf(s1.w - mxv - r1.w * r1.w * dinv) * inv);
    af[8]  = (bf16)(__expf(s2.x - mxv - r2.x * r2.x * dinv) * inv);
    af[9]  = (bf16)(__expf(s2.y - mxv - r2.y * r2.y * dinv) * inv);
    af[10] = (bf16)(__expf(s2.z - mxv - r2.z * r2.z * dinv) * inv);
    af[11] = (bf16)(__expf(s2.w - mxv - r2.w * r2.w * dinv) * inv);
    af[12] = (bf16)(__expf(s3.x - mxv - r3.x * r3.x * dinv) * inv);
    af[13] = (bf16)(__expf(s3.y - mxv - r3.y * r3.y * dinv) * inv);
    af[14] = (bf16)(__expf(s3.z - mxv - r3.z * r3.z * dinv) * inv);
    af[15] = (bf16)(__expf(s3.w - mxv - r3.w * r3.w * dinv) * inv);
#pragma unroll
    for (int jt = 0; jt < 4; ++jt) {
      bf16x16 bfr;
#pragma unroll
      for (int e = 0; e < 16; ++e) {
        int kidx = k0 + lhalf * 8 + (e & 7) + (e >> 3) * 16;
        bfr[e] = (bf16)qkvx[((size_t)b * NP + kidx) * QKVC + 2 * INNERC + h * DHEAD +
                            jt * 16 + lmod];
      }
      o[jt] = wmma_bf16(af, bfr, o[jt]);
    }
  }
#pragma unroll
  for (int jt = 0; jt < 4; ++jt)
#pragma unroll
    for (int r = 0; r < 8; ++r)
      ps[wave][lhalf * 8 + r][jt * 16 + lmod] = o[jt][r];
  __syncthreads();
  for (int idx = threadIdx.x; idx < 16 * 64; idx += 128) {
    int rr = idx >> 6, cc = idx & 63;
    float v = ps[0][rr][cc] + ps[1][rr][cc] + ps[2][rr][cc] + ps[3][rr][cc];
    ao[((size_t)b * NK + mt * 16 + rr) * INNERC + h * DHEAD + cc] = v;
  }
}

// ---------------- class-token attention (tiny, VALU) ----------------
__global__ __launch_bounds__(32)
void attn_cls(const float* __restrict__ qkvc, const float* __restrict__ qkvk,
              const float* __restrict__ maskp, const float* __restrict__ kmaskp,
              float* __restrict__ aoc) {
  int bh = blockIdx.x, b = bh >> 3, h = bh & 7;
  int lane = threadIdx.x;
  __shared__ float sd[NK];
  const float* cq = qkvc + (size_t)b * QKVC + h * DHEAD;
  float m0v = maskp[(size_t)b * NP];
  for (int j = lane; j < NK; j += 32) {
    const float* kk = qkvk + ((size_t)b * NK + j) * QKVC + INNERC + h * DHEAD;
    float dot = 0.f;
#pragma unroll
    for (int d = 0; d < DHEAD; ++d) dot += cq[d] * kk[d];
    dot *= ATTSCALE;
    if (m0v * kmaskp[b * NK + j] < 0.5f) dot = NEGV;
    sd[j] = dot;
  }
  __syncthreads();
  float mx = -3e38f;
  for (int j = lane; j < NK; j += 32) mx = fmaxf(mx, sd[j]);
#pragma unroll
  for (int off = 16; off >= 1; off >>= 1) mx = fmaxf(mx, __shfl_xor(mx, off, 32));
  float sum = 0.f;
  for (int j = lane; j < NK; j += 32) { float e = __expf(sd[j] - mx); sd[j] = e; sum += e; }
#pragma unroll
  for (int off = 16; off >= 1; off >>= 1) sum += __shfl_xor(sum, off, 32);
  float inv = 1.0f / sum;
  __syncthreads();
  for (int d = lane; d < DHEAD; d += 32) {
    float acc = 0.f;
    for (int j = 0; j < NK; ++j)
      acc += sd[j] * qkvk[((size_t)b * NK + j) * QKVC + 2 * INNERC + h * DHEAD + d];
    aoc[(size_t)b * INNERC + h * DHEAD + d] = acc * inv;
  }
}

// ---------------- masked k_reps writeback ----------------
__global__ __launch_bounds__(256)
void write_kreps(const float* __restrict__ kx, const float* __restrict__ kmaskp,
                 float* __restrict__ out) {
  int idx = blockIdx.x * 256 + threadIdx.x;
  if (idx >= BB * NK * DIMC) return;
  int bk = idx / DIMC;
  out[idx] = (kmaskp[bk] < 0.5f) ? 0.f : kx[idx];
}

// ---------------- host launcher ----------------
extern "C" void kernel_launch(void* const* d_in, const int* in_sizes, int n_in,
                              void* d_out, int out_size, void* d_ws, size_t ws_size,
                              hipStream_t stream) {
  (void)in_sizes; (void)n_in; (void)out_size; (void)ws_size;
  const float* in_x   = (const float*)d_in[0];
  const float* in_kx  = (const float*)d_in[1];
  const float* rd     = (const float*)d_in[2];
  const float* in_cl  = (const float*)d_in[3];
  const float* maskp  = (const float*)d_in[4];
  const float* kmaskp = (const float*)d_in[5];
  const float* ln1_g  = (const float*)d_in[6];
  const float* ln1_b  = (const float*)d_in[7];
  const float* w_qkv  = (const float*)d_in[8];
  const float* w_out  = (const float*)d_in[9];
  const float* b_out  = (const float*)d_in[10];
  const float* ln2_g  = (const float*)d_in[11];
  const float* ln2_b  = (const float*)d_in[12];
  const float* w1     = (const float*)d_in[13];
  const float* b1     = (const float*)d_in[14];
  const float* w2     = (const float*)d_in[15];
  const float* b2     = (const float*)d_in[16];

  float* ws = (float*)d_ws;
  size_t off = 0;
  auto carve = [&](size_t n) { float* p = ws + off; off += n; return p; };
  // row counts padded to the GEMM M-tile (128) so staging never needs guards
  const size_t XSZ  = (size_t)BB * NP * DIMC;   // 8192 rows (already mult of 128)
  const size_t KPAD = 384, CPAD = 128;          // 288 -> 384, 2 -> 128 rows
  float* X     = carve(XSZ);
  float* KX    = carve(KPAD * DIMC);
  float* C     = carve(CPAD * DIMC);
  float* LNX   = carve(XSZ);
  float* LNK   = carve(KPAD * DIMC);
  float* LNC   = carve(CPAD * DIMC);
  float* QKV_X = carve((size_t)BB * NP * QKVC);
  float* QKV_K = carve((size_t)(BB * NK + 16) * QKVC);  // +16 zeroed rows (K-pad 160)
  float* QKV_C = carve(CPAD * QKVC);
  float* AO_X  = carve(XSZ);
  float* AO_K  = carve(KPAD * INNERC);
  float* AO_C  = carve(CPAD * INNERC);
  float* H_X   = carve((size_t)BB * NP * MLPC);  // also aliased by S
  float* H_K   = carve(KPAD * MLPC);
  float* H_C   = carve(CPAD * MLPC);
  float* RMAX  = carve((size_t)BB * NHEADS * NK);
  float* RSUM  = carve((size_t)BB * NHEADS * NK);
  float* S     = H_X;  // S [b*h,144,4096] and MLP hidden never live simultaneously

  hipMemcpyAsync(X,  in_x,  XSZ * sizeof(float), hipMemcpyDeviceToDevice, stream);
  hipMemcpyAsync(KX, in_kx, (size_t)BB * NK * DIMC * sizeof(float),
                 hipMemcpyDeviceToDevice, stream);
  hipMemcpyAsync(C,  in_cl, (size_t)BB * DIMC * sizeof(float),
                 hipMemcpyDeviceToDevice, stream);
  // zero the 16 pad rows of QKV_K (read as V/K rows 144..159 with zero A weights)
  hipMemsetAsync(QKV_K + (size_t)BB * NK * QKVC, 0, (size_t)16 * QKVC * sizeof(float),
                 stream);

  const int XROWS = BB * NP;   // 8192
  const int KROWS = BB * NK;   // 288
  const int CROWS = BB;        // 2
  auto lngrid = [](int rows) { return dim3((rows + 7) / 8); };
  auto ggrid  = [](int M, int N) { return dim3((M + 127) / 128, N / 128); };

  for (int l = 0; l < NDEPTH; ++l) {
    const float* g1 = ln1_g + l * DIMC; const float* bb1 = ln1_b + l * DIMC;
    const float* g2 = ln2_g + l * DIMC; const float* bb2 = ln2_b + l * DIMC;
    const float* wq = w_qkv + (size_t)l * DIMC * QKVC;
    const float* wo = w_out + (size_t)l * INNERC * DIMC;
    const float* bo = b_out + (size_t)l * DIMC;
    const float* w1l = w1 + (size_t)l * DIMC * MLPC;
    const float* b1l = b1 + (size_t)l * MLPC;
    const float* w2l = w2 + (size_t)l * MLPC * DIMC;
    const float* b2l = b2 + (size_t)l * DIMC;
    float dinv = 1.0f / (float)(64 << l);   // 1/(2*NPK*2^l)

    // pre-norm replaces the residual stream (matches reference)
    ln_kernel<<<lngrid(XROWS), 256, 0, stream>>>(X,  X,  g1, bb1, XROWS);
    ln_kernel<<<lngrid(KROWS), 256, 0, stream>>>(KX, KX, g1, bb1, KROWS);
    ln_kernel<<<lngrid(CROWS), 256, 0, stream>>>(C,  C,  g1, bb1, CROWS);

    gemm_kernel<<<ggrid(XROWS, QKVC), 256, 0, stream>>>(X,  wq, nullptr, nullptr, QKV_X, XROWS, QKVC, DIMC, 0);
    gemm_kernel<<<ggrid(KROWS, QKVC), 256, 0, stream>>>(KX, wq, nullptr, nullptr, QKV_K, KROWS, QKVC, DIMC, 0);
    gemm_kernel<<<ggrid(CROWS, QKVC), 256, 0, stream>>>(C,  wq, nullptr, nullptr, QKV_C, CROWS, QKVC, DIMC, 0);

    attn_pk_kernel<<<dim3(NP / 32, BB * NHEADS), 64, 0, stream>>>(
        QKV_X, QKV_K, maskp, kmaskp, rd, AO_X, dinv);

    attn_kp_dots<<<dim3(NP / 64, NK / 16, BB * NHEADS), 32, 0, stream>>>(
        QKV_X, QKV_K, maskp, kmaskp, S);
    attn_kp_stats<<<dim3(BB * NHEADS * NK), 256, 0, stream>>>(S, RMAX, RSUM);
    attn_kp_av<<<dim3(NK / 16, BB * NHEADS), 128, 0, stream>>>(
        S, RMAX, RSUM, rd, QKV_X, AO_K, dinv);

    attn_cls<<<dim3(BB * NHEADS), 32, 0, stream>>>(QKV_C, QKV_K, maskp, kmaskp, AO_C);

    // out-proj + bias + residual (in place on residual stream)
    gemm_kernel<<<ggrid(XROWS, DIMC), 256, 0, stream>>>(AO_X, wo, bo, X,  X,  XROWS, DIMC, INNERC, 0);
    gemm_kernel<<<ggrid(KROWS, DIMC), 256, 0, stream>>>(AO_K, wo, bo, KX, KX, KROWS, DIMC, INNERC, 0);
    gemm_kernel<<<ggrid(CROWS, DIMC), 256, 0, stream>>>(AO_C, wo, bo, C,  C,  CROWS, DIMC, INNERC, 0);

    // MLP: ln2 -> w1+gelu -> w2 + residual
    ln_kernel<<<lngrid(XROWS), 256, 0, stream>>>(X,  LNX, g2, bb2, XROWS);
    ln_kernel<<<lngrid(KROWS), 256, 0, stream>>>(KX, LNK, g2, bb2, KROWS);
    ln_kernel<<<lngrid(CROWS), 256, 0, stream>>>(C,  LNC, g2, bb2, CROWS);

    gemm_kernel<<<ggrid(XROWS, MLPC), 256, 0, stream>>>(LNX, w1l, b1l, nullptr, H_X, XROWS, MLPC, DIMC, 1);
    gemm_kernel<<<ggrid(KROWS, MLPC), 256, 0, stream>>>(LNK, w1l, b1l, nullptr, H_K, KROWS, MLPC, DIMC, 1);
    gemm_kernel<<<ggrid(CROWS, MLPC), 256, 0, stream>>>(LNC, w1l, b1l, nullptr, H_C, CROWS, MLPC, DIMC, 1);

    gemm_kernel<<<ggrid(XROWS, DIMC), 256, 0, stream>>>(H_X, w2l, b2l, X,  X,  XROWS, DIMC, MLPC, 0);
    gemm_kernel<<<ggrid(KROWS, DIMC), 256, 0, stream>>>(H_K, w2l, b2l, KX, KX, KROWS, DIMC, MLPC, 0);
    gemm_kernel<<<ggrid(CROWS, DIMC), 256, 0, stream>>>(H_C, w2l, b2l, C,  C,  CROWS, DIMC, MLPC, 0);

    write_kreps<<<dim3((BB * NK * DIMC + 255) / 256), 256, 0, stream>>>(
        KX, kmaskp, (float*)d_out + (size_t)l * BB * NK * DIMC);
  }

  // clst output after the stacked k_reps
  hipMemcpyAsync((float*)d_out + (size_t)NDEPTH * BB * NK * DIMC, C,
                 (size_t)BB * DIMC * sizeof(float), hipMemcpyDeviceToDevice, stream);
}
